// SparseAttention_22840636080900
// MI455X (gfx1250) — matrix operand order repaired and checked by reading, not compiled, not run on hardware
//
#include <hip/hip_runtime.h>
#include <hip/hip_bf16.h>

#define Q_N   65536
#define KV_N  65536
#define C     256
#define H     8
#define N_E   524288
#define SCALE 0.0625f     // 1/sqrt(256)
#define LN_EPS 1e-5f

typedef _Float16 f16;
typedef __attribute__((ext_vector_type(16))) _Float16 v16h;
typedef __attribute__((ext_vector_type(8)))  _Float16 v8h;
typedef __attribute__((ext_vector_type(8)))  float    v8f;
typedef __attribute__((ext_vector_type(4)))  float    v4f;

union HFrag { v16h v; v8h h8[2]; };

// A fragment, 16-bit A 16x32 layout (ISA 7.12.2):
// lane-half h: halves 0-7 = cols [kb+8h, +8), halves 8-15 = cols [kb+16+8h, +8)
__device__ __forceinline__ v16h load_frag_a(const f16* __restrict__ row, int kb, int h)
{
    HFrag f;
    f.h8[0] = *(const v8h*)(row + kb + 8 * h);
    f.h8[1] = *(const v8h*)(row + kb + 16 + 8 * h);
    return f.v;
}
// B fragment, 16-bit B 32x16 layout: lane-half h holds K = [kb+8h, +8) and [kb+16+8h, +8) for col N.
// B[k][n] = W[n][k]  =>  two runs of 8 contiguous f16 from W row n.
__device__ __forceinline__ v16h load_frag_b(const f16* __restrict__ wrow, int kb, int h)
{
    HFrag f;
    f.h8[0] = *(const v8h*)(wrow + kb + 8 * h);
    f.h8[1] = *(const v8h*)(wrow + kb + 16 + 8 * h);
    return f.v;
}

// ---------------------------------------------------------------------------
// LayerNorm (f32 math) with f16 output. One wave per row; lane owns the 8
// contiguous channels [8*lane, 8*lane+8) -> float4 x2 loads, one b128 store.
// ---------------------------------------------------------------------------
__global__ __launch_bounds__(256) void ln_f16_kernel(
    const float* __restrict__ x, const float* __restrict__ w,
    const float* __restrict__ b, f16* __restrict__ y, int M)
{
    int row  = blockIdx.x * 8 + (threadIdx.x >> 5);
    int lane = threadIdx.x & 31;
    if (row >= M) return;
    const float* xr = x + (size_t)row * C + 8 * lane;

    v4f x0 = *(const v4f*)(xr);
    v4f x1 = *(const v4f*)(xr + 4);
    float v[8] = { x0[0], x0[1], x0[2], x0[3], x1[0], x1[1], x1[2], x1[3] };

    float s = 0.f;
#pragma unroll
    for (int j = 0; j < 8; ++j) s += v[j];
#pragma unroll
    for (int m = 1; m < 32; m <<= 1) s += __shfl_xor(s, m, 32);
    float mean = s * (1.f / C);

    float vs = 0.f;
#pragma unroll
    for (int j = 0; j < 8; ++j) { float d = v[j] - mean; vs += d * d; }
#pragma unroll
    for (int m = 1; m < 32; m <<= 1) vs += __shfl_xor(vs, m, 32);
    float rstd = rsqrtf(vs * (1.f / C) + LN_EPS);

    v8h o;
#pragma unroll
    for (int j = 0; j < 8; ++j) {
        int c = 8 * lane + j;
        o[j] = (f16)((v[j] - mean) * rstd * w[c] + b[c]);
    }
    *(v8h*)(y + (size_t)row * C + 8 * lane) = o;
}

// ---------------------------------------------------------------------------
// f32 -> f16 conversion, 8 elements per thread
// ---------------------------------------------------------------------------
__global__ __launch_bounds__(256) void conv_f16_kernel(
    const float* __restrict__ x, f16* __restrict__ y, int n8)
{
    int i = blockIdx.x * 256 + threadIdx.x;
    if (i >= n8) return;
    v4f a = ((const v4f*)x)[2 * i];
    v4f b = ((const v4f*)x)[2 * i + 1];
    v8h o = { (f16)a[0], (f16)a[1], (f16)a[2], (f16)a[3],
              (f16)b[0], (f16)b[1], (f16)b[2], (f16)b[3] };
    ((v8h*)y)[i] = o;
}

// ---------------------------------------------------------------------------
// GEMM  out[M,C] = A[M,C](f16) @ W16[C,C]^T + bv (+ resid), f32 accumulate,
// via V_WMMA_F32_16X16X32_F16. Wave computes 32(M) x 64(N): 2 A fragments
// shared against 4 B fragments -> 8 wmmas per 12 b128 loads per K-step.
// K loop (C/32 = 8 steps) is fully unrolled: fragments land in fresh virtual
// registers (no cross-iteration copies -> no WMMA WAR-hazard v_nops) and the
// scheduler clause-groups/hoists the loads across the WMMA stream.
// Block = 4 waves = 4 N-strips of the same 32 rows.
// ---------------------------------------------------------------------------
template <typename OutT>
__global__ __launch_bounds__(128) void gemm_f16_wmma(
    const f16* __restrict__ A, const f16* __restrict__ W16,
    const float* __restrict__ bv, const float* __restrict__ resid,
    OutT* __restrict__ out, int M)
{
    int lane  = threadIdx.x & 31;
    int nBase = (threadIdx.x >> 5) * 64;
    int mBase = blockIdx.x * 32;
    if (mBase >= M) return;
    int h = lane >> 4;
    int l = lane & 15;

    const f16* aRow0 = A + (size_t)(mBase + l) * C;
    const f16* aRow1 = A + (size_t)(mBase + 16 + l) * C;
    const f16* wRow[4];
#pragma unroll
    for (int t = 0; t < 4; ++t) wRow[t] = W16 + (size_t)(nBase + t * 16 + l) * C;

    v8f accA[4] = {{0.f}, {0.f}, {0.f}, {0.f}};
    v8f accB[4] = {{0.f}, {0.f}, {0.f}, {0.f}};

#pragma unroll
    for (int kb = 0; kb < C; kb += 32) {
        v16h a0 = load_frag_a(aRow0, kb, h);
        v16h a1 = load_frag_a(aRow1, kb, h);
#pragma unroll
        for (int t = 0; t < 4; ++t) {
            v16h bfrag = load_frag_b(wRow[t], kb, h);
            accA[t] = __builtin_amdgcn_wmma_f32_16x16x32_f16(
                false, a0, false, bfrag, (short)0, accA[t], false, false);
            accB[t] = __builtin_amdgcn_wmma_f32_16x16x32_f16(
                false, a1, false, bfrag, (short)0, accB[t], false, false);
        }
    }

    // D layout: acc[t][r] -> (M = mBase (+16) + r + 8h, N = nBase + t*16 + l)
#pragma unroll
    for (int t = 0; t < 4; ++t) {
        int n = nBase + t * 16 + l;
        float bn = bv[n];
#pragma unroll
        for (int r = 0; r < 8; ++r) {
            int m0 = mBase + r + 8 * h;
            int m1 = m0 + 16;
            float d0 = accA[t][r] + bn;
            float d1 = accB[t][r] + bn;
            if (resid) {
                d0 += resid[(size_t)m0 * C + n];
                d1 += resid[(size_t)m1 * C + n];
            }
            out[(size_t)m0 * C + n] = (OutT)d0;
            out[(size_t)m1 * C + n] = (OutT)d1;
        }
    }
}

// ---------------------------------------------------------------------------
// Edge similarity: one wave per edge; lane owns channels [8*lane, +8) so
// head(lane) = lane>>2 and per-head dot is a 2-step 4-lane butterfly.
// f16 gathered rows (half the L2 traffic), f32 math.
// ---------------------------------------------------------------------------
__global__ __launch_bounds__(256) void edge_sim_kernel(
    const f16* __restrict__ qp, const f16* __restrict__ kp,
    const float* __restrict__ bias, const int* __restrict__ src,
    const int* __restrict__ dst, float* __restrict__ sim)
{
    int e    = blockIdx.x * 8 + (threadIdx.x >> 5);
    int lane = threadIdx.x & 31;
    if (e >= N_E) return;

    v8h qv = *(const v8h*)(qp + (size_t)dst[e] * C + 8 * lane);
    v8h kv = *(const v8h*)(kp + (size_t)src[e] * C + 8 * lane);

    float p = 0.f;
#pragma unroll
    for (int i = 0; i < 8; ++i) p += (float)qv[i] * (float)kv[i];
    p += __shfl_xor(p, 1, 32);
    p += __shfl_xor(p, 2, 32);

    if ((lane & 3) == 0) {
        int hd = lane >> 2;
        sim[(size_t)e * H + hd] = (p + bias[(size_t)e * H + hd]) * SCALE;
    }
}

// ---------------------------------------------------------------------------
// Segmented softmax + weighted-V accumulation (dst sorted). One wave/query.
// simw holds sim on entry; overwritten in place with final weights (output).
// attn written as f16 for the final WMMA GEMM.
// ---------------------------------------------------------------------------
__device__ __forceinline__ int lower_bound_i32(const int* __restrict__ a, int n, int key)
{
    int lo = 0, hi = n;
    while (lo < hi) {
        int mid = (lo + hi) >> 1;
        if (a[mid] < key) lo = mid + 1; else hi = mid;
    }
    return lo;
}

__global__ __launch_bounds__(256) void attn_kernel(
    const f16* __restrict__ kp, const int* __restrict__ src,
    const int* __restrict__ dst, float* __restrict__ simw,
    f16* __restrict__ attn)
{
    int qi   = blockIdx.x * 8 + (threadIdx.x >> 5);
    int lane = threadIdx.x & 31;
    if (qi >= Q_N) return;

    int lo = lower_bound_i32(dst, N_E, qi);
    int hi = lower_bound_i32(dst, N_E, qi + 1);
    int myh = lane >> 2;

    float m[8];
#pragma unroll
    for (int j = 0; j < 8; ++j) m[j] = -__builtin_inff();
    for (int e = lo; e < hi; ++e) {
#pragma unroll
        for (int j = 0; j < 8; ++j) m[j] = fmaxf(m[j], simw[(size_t)e * H + j]);
    }

    float s[8];
#pragma unroll
    for (int j = 0; j < 8; ++j) s[j] = 0.f;
    for (int e = lo; e < hi; ++e) {
#pragma unroll
        for (int j = 0; j < 8; ++j) s[j] += __expf(simw[(size_t)e * H + j] - m[j]);
    }
    float inv[8];
#pragma unroll
    for (int j = 0; j < 8; ++j) inv[j] = 1.f / (1e-8f + s[j]);

    float acc[8];
#pragma unroll
    for (int j = 0; j < 8; ++j) acc[j] = 0.f;

    for (int e = lo; e < hi; ++e) {
        float w[8];
#pragma unroll
        for (int j = 0; j < 8; ++j)
            w[j] = __expf(simw[(size_t)e * H + j] - m[j]) * inv[j];

        // finalize weights output (in-place over sim; segment is wave-private)
        float wst = 0.f;
#pragma unroll
        for (int j = 0; j < 8; ++j) if (lane == j) wst = w[j];
        if (lane < 8) simw[(size_t)e * H + lane] = wst;

        float wm = 0.f;
#pragma unroll
        for (int j = 0; j < 8; ++j) if (myh == j) wm = w[j];

        v8h kv = *(const v8h*)(kp + (size_t)src[e] * C + 8 * lane);
#pragma unroll
        for (int i = 0; i < 8; ++i) acc[i] += wm * (float)kv[i];
    }

    v8h o;
#pragma unroll
    for (int i = 0; i < 8; ++i) o[i] = (f16)acc[i];
    *(v8h*)(attn + (size_t)qi * C + 8 * lane) = o;
}

// ---------------------------------------------------------------------------
// Launch sequence
// ---------------------------------------------------------------------------
extern "C" void kernel_launch(void* const* d_in, const int* in_sizes, int n_in,
                              void* d_out, int out_size, void* d_ws, size_t ws_size,
                              hipStream_t stream)
{
    const float* q    = (const float*)d_in[0];
    const float* k    = (const float*)d_in[1];
    const float* bias = (const float*)d_in[2];
    const float* ln_w = (const float*)d_in[3];
    const float* ln_b = (const float*)d_in[4];
    const float* wq   = (const float*)d_in[5];
    const float* bq   = (const float*)d_in[6];
    const float* wk   = (const float*)d_in[7];
    const float* bk   = (const float*)d_in[8];
    const float* wo   = (const float*)d_in[9];
    const float* bo   = (const float*)d_in[10];
    const int*   src  = (const int*)d_in[11];
    const int*   dst  = (const int*)d_in[12];

    float* out     = (float*)d_out;                    // [Q_N, C]
    float* weights = out + (size_t)Q_N * C;            // [N_E, H] (sim scratch, then weights)

    const size_t MAT = (size_t)Q_N * C;                // 16,777,216 elems
    f16* h     = (f16*)d_ws;
    f16* qn16  = h;                                    // [Q_N, C]
    f16* k16   = h + MAT;                              // [KV_N, C]
    f16* qp16  = h + 2 * MAT;                          // [Q_N, C]
    f16* kp16  = h + 3 * MAT;                          // [KV_N, C]
    f16* wq16  = h + 4 * MAT;                          // [C, C]
    f16* wk16  = wq16 + (size_t)C * C;
    f16* wo16  = wk16 + (size_t)C * C;
    f16* attn16 = qn16;                                // reuse: qn16 dead after GEMM1

    // 1. pre-LN (f32 math) -> f16
    ln_f16_kernel<<<Q_N / 8, 256, 0, stream>>>(q, ln_w, ln_b, qn16, Q_N);
    // 2. f16 conversions: k and the three weight matrices
    conv_f16_kernel<<<(KV_N * C / 8) / 256, 256, 0, stream>>>(k, k16, KV_N * C / 8);
    conv_f16_kernel<<<(C * C / 8 + 255) / 256, 256, 0, stream>>>(wq, wq16, C * C / 8);
    conv_f16_kernel<<<(C * C / 8 + 255) / 256, 256, 0, stream>>>(wk, wk16, C * C / 8);
    conv_f16_kernel<<<(C * C / 8 + 255) / 256, 256, 0, stream>>>(wo, wo16, C * C / 8);
    // 3. qp = LN(q) @ wq^T + bq   (f16 out for cheap gathers)
    gemm_f16_wmma<f16><<<Q_N / 32, 128, 0, stream>>>(qn16, wq16, bq, nullptr, qp16, Q_N);
    // 4. kp = k @ wk^T + bk       (v = kp, same_kv)
    gemm_f16_wmma<f16><<<KV_N / 32, 128, 0, stream>>>(k16, wk16, bk, nullptr, kp16, KV_N);
    // 5. per-edge sim -> weights region (scratch)
    edge_sim_kernel<<<N_E / 8, 256, 0, stream>>>(qp16, kp16, bias, src, dst, weights);
    // 6. segmented softmax + AV; weights finalized in place, attn -> f16
    attn_kernel<<<Q_N / 8, 256, 0, stream>>>(kp16, src, dst, weights, attn16);
    // 7. out = attn @ wo^T + bo + residual(q)   (f32 out)
    gemm_f16_wmma<float><<<Q_N / 32, 128, 0, stream>>>(attn16, wo16, bo, q, out, Q_N);
}
